// SAGEConvMLPModel_17712445128821
// MI455X (gfx1250) — matrix operand order repaired and checked by reading, not compiled, not run on hardware
//
#include <hip/hip_runtime.h>
#include <hip/hip_bf16.h>

typedef __attribute__((ext_vector_type(16))) _Float16     v16h;
typedef __attribute__((ext_vector_type(8)))  _Float16     v8h;
typedef __attribute__((ext_vector_type(8)))  float        v8f;
typedef __attribute__((ext_vector_type(4)))  unsigned int v4u;
typedef __attribute__((ext_vector_type(8)))  int          v8i;
typedef __attribute__((ext_vector_type(4)))  int          v4i;

#define IN_CH   1024
#define FC_CH   128
#define NCOLS   256      // [Wl out | Wr out]
#define ADD_CH  20
#define CAT_CH  148      // FC_CH + ADD_CH
#define MID_CH  37
#define OUT_CH  3
#define EPSBN   1e-5f
#define NSLOPE  0.01f
#define KCHUNK  64       // K elems staged per TDM load (16 x 64 f32 tile)
#define ROWPAD  4        // LDS pad DWORDs inserted by TDM after each 64-DWORD row
#define ROWSTR  (KCHUNK + ROWPAD)   // 68-float LDS row stride -> bank shift of 4/row

// ---------------------------------------------------------------------------
// Kernel 0: convert [Wl (128x1024); Wr (128x1024)] f32 -> f16, row-major over K
// ---------------------------------------------------------------------------
__global__ void convert_weights(const float* __restrict__ Wl,
                                const float* __restrict__ Wr,
                                _Float16* __restrict__ Wh) {
    int i = blockIdx.x * 256 + threadIdx.x;          // 0 .. 256*1024-1
    int n = i >> 10;
    int k = i & 1023;
    float v = (n < FC_CH) ? Wl[n * IN_CH + k] : Wr[(n - FC_CH) * IN_CH + k];
    Wh[i] = (_Float16)v;
}

// ---------------------------------------------------------------------------
// TDM helper: DMA a [16 x KCHUNK] f32 tile of `feat` (row stride IN_CH) into
// LDS at byte offset lds_addr, inserting ROWPAD DWORDs of LDS padding after
// every 64-DWORD row (pad_interval=5 -> 64 DW, pad_amount=3 -> 4 DW) so rows
// land on different LDS banks. Rows beyond `rows_remaining` are OOB -> zero.
// Descriptor per CDNA5 ISA.
// ---------------------------------------------------------------------------
__device__ __forceinline__ void tdm_load_tile(const float* feat_tile_start,
                                              unsigned int lds_addr,
                                              unsigned int rows_remaining) {
    unsigned long long ga = (unsigned long long)(const void*)feat_tile_start;
    v4u g0;
    g0[0] = 1u;                                   // count=1, user descriptor
    g0[1] = lds_addr;                             // LDS byte address
    g0[2] = (unsigned int)ga;                     // global_addr[31:0]
    g0[3] = (unsigned int)((ga >> 32) & 0x1FFFFFFu) | (2u << 30); // addr[56:32] | type=2
    v8i g1;
    g1[0] = (2 << 16)                             // data_size=2 -> 4-byte elems
          | (1 << 20)                             // pad_enable
          | (5 << 22)                             // pad_interval: 64 DWORDs
          | (3 << 25);                            // pad_amount: 4 DWORDs
    g1[1] = (int)((IN_CH & 0xFFFF) << 16);        // tensor_dim0[15:0]
    g1[2] = (int)(((IN_CH >> 16) & 0xFFFF) |
                  ((rows_remaining & 0xFFFF) << 16));  // dim0 hi | tensor_dim1 lo
    g1[3] = (int)(((rows_remaining >> 16) & 0xFFFF) |
                  (KCHUNK << 16));                // tensor_dim1 hi | tile_dim0
    g1[4] = 16;                                   // tile_dim1=16 rows (tile_dim2=0)
    g1[5] = IN_CH;                                // tensor_dim0_stride[31:0]
    g1[6] = 0;                                    // stride0 hi | stride1 lo (unused, 2D)
    g1[7] = 0;
    v4i z4 = {0, 0, 0, 0};
#if defined(__clang_major__) && __clang_major__ >= 23
    v8i z8 = {0, 0, 0, 0, 0, 0, 0, 0};
    __builtin_amdgcn_tensor_load_to_lds(g0, g1, z4, z4, z8, 0);
#else
    __builtin_amdgcn_tensor_load_to_lds(g0, g1, z4, z4, 0);
#endif
}

// ---------------------------------------------------------------------------
// Kernel 1: YZ[N x 256] = features[N x 1024] @ Wh^T   (WMMA f32_16x16x32_f16)
// Block = 256 threads = 8 waves. Block tile: M=16, N=256. Wave tile: 16x32.
// A tiles double-buffered into LDS by the Tensor Data Mover (wave 0 issues).
// ---------------------------------------------------------------------------
__global__ void gemm_wmma_f16(const float* __restrict__ feat,
                              const _Float16* __restrict__ Wh,
                              float* __restrict__ YZ, int nrows) {
    __shared__ __align__(16) float Abuf[2][16 * ROWSTR];   // 2 x padded f32 tiles

    const int tid   = threadIdx.x;
    const int wave  = tid >> 5;
    const int lane  = tid & 31;
    const int half_ = lane >> 4;    // 0: lanes 0-15, 1: lanes 16-31
    const int l15   = lane & 15;
    const int m0    = blockIdx.x * 16;
    const int n0    = wave * 32;
    const unsigned int rem = (unsigned int)(nrows - m0);   // rows from tile start

    v8f acc0 = {};
    v8f acc1 = {};

    // Prologue: wave 0 kicks off the first tile DMA.
    if (tid < 32) {
        tdm_load_tile(feat + (size_t)m0 * IN_CH,
                      (unsigned int)(unsigned long long)&Abuf[0][0], rem);
    }

    for (int kk = 0; kk < IN_CH; kk += KCHUNK) {
        const int cur = (kk >> 6) & 1;
        if (tid < 32) {
            if (kk + KCHUNK < IN_CH) {
                // Issue next tile into the other buffer, then wait until only
                // that one is outstanding (TDM completes in-order per wave).
                tdm_load_tile(feat + (size_t)m0 * IN_CH + kk + KCHUNK,
                              (unsigned int)(unsigned long long)&Abuf[cur ^ 1][0], rem);
                __builtin_amdgcn_s_wait_tensorcnt(1);
            } else {
                __builtin_amdgcn_s_wait_tensorcnt(0);
            }
        }
        __syncthreads();            // Abuf[cur] ready for all waves

        const float* arow = &Abuf[cur][l15 * ROWSTR];
        #pragma unroll
        for (int ks = 0; ks < KCHUNK; ks += 32) {
            // A fragment (ISA layout): elems 0-7 -> k=half*8+j, 8-15 -> 16+half*8+j
            const float4 f0 = *(const float4*)&arow[ks + half_ * 8];
            const float4 f1 = *(const float4*)&arow[ks + half_ * 8 + 4];
            const float4 f2 = *(const float4*)&arow[ks + 16 + half_ * 8];
            const float4 f3 = *(const float4*)&arow[ks + 16 + half_ * 8 + 4];
            v16h a;
            a[0]  = (_Float16)f0.x; a[1]  = (_Float16)f0.y;
            a[2]  = (_Float16)f0.z; a[3]  = (_Float16)f0.w;
            a[4]  = (_Float16)f1.x; a[5]  = (_Float16)f1.y;
            a[6]  = (_Float16)f1.z; a[7]  = (_Float16)f1.w;
            a[8]  = (_Float16)f2.x; a[9]  = (_Float16)f2.y;
            a[10] = (_Float16)f2.z; a[11] = (_Float16)f2.w;
            a[12] = (_Float16)f3.x; a[13] = (_Float16)f3.y;
            a[14] = (_Float16)f3.z; a[15] = (_Float16)f3.w;

            // B fragments: lane n, elems 0-7 -> k=half*8+j, 8-15 -> 16+half*8+j
            const v8h* bp0 = (const v8h*)&Wh[(size_t)(n0 + l15) * IN_CH + kk + ks + half_ * 8];
            const v8h* bp1 = (const v8h*)&Wh[(size_t)(n0 + 16 + l15) * IN_CH + kk + ks + half_ * 8];
            v8h b0l = bp0[0], b0h = bp0[2];
            v8h b1l = bp1[0], b1h = bp1[2];
            v16h b0, b1;
            #pragma unroll
            for (int j = 0; j < 8; ++j) {
                b0[j] = b0l[j]; b0[8 + j] = b0h[j];
                b1[j] = b1l[j]; b1[8 + j] = b1h[j];
            }

            acc0 = __builtin_amdgcn_wmma_f32_16x16x32_f16(false, a, false, b0,
                                                          (short)0, acc0, false, false);
            acc1 = __builtin_amdgcn_wmma_f32_16x16x32_f16(false, a, false, b1,
                                                          (short)0, acc1, false, false);
        }
        __syncthreads();            // all reads of Abuf[cur] done before reuse
    }

    // C/D layout: VGPR r -> row (r + half*8), col = l15
    #pragma unroll
    for (int r = 0; r < 8; ++r) {
        int row = m0 + r + half_ * 8;
        if (row < nrows) {
            YZ[(size_t)row * NCOLS + n0 + l15]      = acc0[r];
            YZ[(size_t)row * NCOLS + n0 + 16 + l15] = acc1[r];
        }
    }
}

// ---------------------------------------------------------------------------
// Kernel 2: edge scatter. One wave per edge; lane handles 4 channels of Y=YZ[:, :128].
// ACC[dst] += Y[src]; DEG[dst] += 1.
// ---------------------------------------------------------------------------
__global__ void edge_scatter(const int* __restrict__ edges,
                             const float* __restrict__ YZ,
                             float* __restrict__ acc,
                             float* __restrict__ deg, int nedges) {
    int gid  = blockIdx.x * blockDim.x + threadIdx.x;
    int e    = gid >> 5;
    int lane = gid & 31;
    if (e >= nedges) return;
    int src = edges[e];
    int dst = edges[nedges + e];
    const float4 v = *(const float4*)&YZ[(size_t)src * NCOLS + lane * 4];
    float* ap = &acc[(size_t)dst * FC_CH + lane * 4];
    unsafeAtomicAdd(ap + 0, v.x);
    unsafeAtomicAdd(ap + 1, v.y);
    unsafeAtomicAdd(ap + 2, v.z);
    unsafeAtomicAdd(ap + 3, v.w);
    if (lane == 0) unsafeAtomicAdd(&deg[dst], 1.0f);
}

// ---------------------------------------------------------------------------
// Kernel 3: per-node fused epilogue. One wave per node, 8 nodes per block.
// x = ACC/deg + bl + Z, leaky-relu, concat add-feats -> fc1 -> relu -> BN -> fc2
// ---------------------------------------------------------------------------
__global__ void node_mlp(const float* __restrict__ YZ,
                         const float* __restrict__ acc,
                         const float* __restrict__ deg,
                         const float* __restrict__ addf,
                         const float* __restrict__ bl,
                         const float* __restrict__ W1, const float* __restrict__ b1,
                         const float* __restrict__ W2, const float* __restrict__ b2,
                         const float* __restrict__ gamma, const float* __restrict__ beta,
                         const float* __restrict__ rmean, const float* __restrict__ rvar,
                         float* __restrict__ out, int nnodes) {
    __shared__ float sW1[MID_CH * 152];     // W1 staged once per block (padded rows)
    __shared__ float hbuf[8][CAT_CH + 4];   // per-wave concat vector
    __shared__ float midbuf[8][MID_CH + 3]; // per-wave fc1 output

    const int tid  = threadIdx.x;
    const int wave = tid >> 5;
    const int lane = tid & 31;
    const int n    = blockIdx.x * 8 + wave;

    for (int i = tid; i < MID_CH * CAT_CH; i += 256) {
        int r = i / CAT_CH, c = i - r * CAT_CH;
        sW1[r * 152 + c] = W1[i];
    }

    if (n < nnodes) {
        float inv = 1.0f / fmaxf(deg[n], 1.0f);
        const float4 a  = *(const float4*)&acc[(size_t)n * FC_CH + lane * 4];
        const float4 z  = *(const float4*)&YZ[(size_t)n * NCOLS + FC_CH + lane * 4];
        const float4 bb = *(const float4*)&bl[lane * 4];
        float x0 = a.x * inv + bb.x + z.x;
        float x1 = a.y * inv + bb.y + z.y;
        float x2 = a.z * inv + bb.z + z.z;
        float x3 = a.w * inv + bb.w + z.w;
        hbuf[wave][lane * 4 + 0] = (x0 >= 0.f) ? x0 : NSLOPE * x0;
        hbuf[wave][lane * 4 + 1] = (x1 >= 0.f) ? x1 : NSLOPE * x1;
        hbuf[wave][lane * 4 + 2] = (x2 >= 0.f) ? x2 : NSLOPE * x2;
        hbuf[wave][lane * 4 + 3] = (x3 >= 0.f) ? x3 : NSLOPE * x3;
        if (lane < ADD_CH)
            hbuf[wave][FC_CH + lane] = addf[(size_t)n * ADD_CH + lane];
    }
    __syncthreads();

    if (n < nnodes) {
        #pragma unroll
        for (int t = 0; t < 2; ++t) {
            int o = lane + t * 32;
            if (o < MID_CH) {
                float s = b1[o];
                const float* wr = &sW1[o * 152];
                const float* hv = hbuf[wave];
                #pragma unroll 4
                for (int k = 0; k < CAT_CH; ++k) s += wr[k] * hv[k];
                s = fmaxf(s, 0.0f);                                    // relu
                s = gamma[o] * (s - rmean[o]) * rsqrtf(rvar[o] + EPSBN) + beta[o];
                midbuf[wave][o] = s;
            }
        }
    }
    __syncthreads();

    if (n < nnodes && lane < OUT_CH) {
        float s = b2[lane];
        #pragma unroll
        for (int k = 0; k < MID_CH; ++k) s += W2[lane * MID_CH + k] * midbuf[wave][k];
        out[(size_t)n * OUT_CH + lane] = s;
    }
}

// ---------------------------------------------------------------------------
extern "C" void kernel_launch(void* const* d_in, const int* in_sizes, int n_in,
                              void* d_out, int out_size, void* d_ws, size_t ws_size,
                              hipStream_t stream) {
    const float* features = (const float*)d_in[0];
    const int*   edges    = (const int*)d_in[1];
    // d_in[2] (edges2), d_in[3] (edge_features) unused by the reference output
    const float* addf  = (const float*)d_in[4];
    const float* Wl    = (const float*)d_in[5];
    const float* bl    = (const float*)d_in[6];
    const float* Wr    = (const float*)d_in[7];
    const float* W1    = (const float*)d_in[8];
    const float* b1    = (const float*)d_in[9];
    const float* W2    = (const float*)d_in[10];
    const float* b2    = (const float*)d_in[11];
    const float* gamma = (const float*)d_in[12];
    const float* beta  = (const float*)d_in[13];
    const float* rmean = (const float*)d_in[14];
    const float* rvar  = (const float*)d_in[15];
    float* out = (float*)d_out;

    const int N = in_sizes[0] / IN_CH;      // 25000
    const int E = in_sizes[1] / 2;          // 400000

    // Workspace layout (256B aligned slices)
    char* ws = (char*)d_ws;
    size_t off = 0;
    float* YZ  = (float*)(ws + off); off += ((size_t)N * NCOLS * 4 + 255) & ~(size_t)255;
    float* ACC = (float*)(ws + off); off += ((size_t)N * FC_CH * 4 + 255) & ~(size_t)255;
    float* DEG = (float*)(ws + off); off += ((size_t)N * 4 + 255) & ~(size_t)255;
    _Float16* Wh = (_Float16*)(ws + off); off += (size_t)NCOLS * IN_CH * 2;

    // Zero accumulators every launch (ws is not re-poisoned between replays)
    hipMemsetAsync(ACC, 0, (size_t)N * FC_CH * 4, stream);
    hipMemsetAsync(DEG, 0, (size_t)N * 4, stream);

    convert_weights<<<(NCOLS * IN_CH) / 256, 256, 0, stream>>>(Wl, Wr, Wh);

    gemm_wmma_f16<<<(N + 15) / 16, 256, 0, stream>>>(features, Wh, YZ, N);

    edge_scatter<<<(E * 32 + 255) / 256, 256, 0, stream>>>(edges, YZ, ACC, DEG, E);

    node_mlp<<<(N + 7) / 8, 256, 0, stream>>>(YZ, ACC, DEG, addf, bl,
                                              W1, b1, W2, b2,
                                              gamma, beta, rmean, rvar, out, N);
}
